// MultiHeadAttention_64613488001115
// MI455X (gfx1250) — compile-verified, matrix operand order checked by reading, not run on hardware
//
#include <hip/hip_runtime.h>

// ---------------------------------------------------------------------------
// MI455X (gfx1250) causal multi-head attention:
//   q/k/v projections (TDM-staged LDS WMMA GEMM) + flash attention +
//   output projection. All matmuls: v_wmma_f32_16x16x32_bf16.
// ---------------------------------------------------------------------------

#define D_MODEL 1024
#define HEADS   16
#define DH      64
#define BATCH   2
#define SEQ     2048
#define NTOK    (BATCH * SEQ)   // 4096

typedef __bf16        v16bf __attribute__((ext_vector_type(16)));
typedef float         v8f   __attribute__((ext_vector_type(8)));
typedef unsigned int  u32x4 __attribute__((ext_vector_type(4)));
typedef unsigned int  tdmg0 __attribute__((ext_vector_type(4)));
typedef int           tdmg1 __attribute__((ext_vector_type(8)));
typedef int           tdmg2 __attribute__((ext_vector_type(4)));

#if __has_builtin(__builtin_amdgcn_tensor_load_to_lds)
#define HAS_TDM 1
#else
#define HAS_TDM 0
#endif

union FragU { v16bf v; u32x4 q[2]; };

// fp32 -> bf16 round-to-nearest-even
__device__ __forceinline__ unsigned short f2bf(float f) {
  unsigned int u = __float_as_uint(f);
  u += 0x7FFFu + ((u >> 16) & 1u);
  return (unsigned short)(u >> 16);
}

// One 16x32 bf16 WMMA operand fragment from a row-major tile.
// ISA 16-bit A layout: lane L<16 -> row L, K {0..7,16..23}; lane L>=16 ->
// row L-16, K {8..15,24..31}. Two b128 loads. B fragments (32x16, lane = N)
// use the same loader on the K-major source of that column.
__device__ __forceinline__ v16bf load_frag(const unsigned short* p, int ld, int lane) {
  const int row = lane & 15;
  const int k0  = (lane >> 4) << 3;   // 0 or 8
  const unsigned short* r0 = p + (size_t)row * ld + k0;
  FragU f;
  f.q[0] = *(const u32x4*)(r0);
  f.q[1] = *(const u32x4*)(r0 + 16);
  return f.v;
}

__device__ __forceinline__ v8f wmma_bf16(v16bf a, v16bf b, v8f c) {
  return __builtin_amdgcn_wmma_f32_16x16x32_bf16(false, a, false, b, (short)0, c,
                                                 false, false);
}

// Row-max across the 16-lane half that holds one C-matrix row.
__device__ __forceinline__ float redmax16(float x) {
#pragma unroll
  for (int m = 1; m < 16; m <<= 1) x = fmaxf(x, __shfl_xor(x, m, 32));
  return x;
}

// LDS byte offset of a generic pointer into shared memory (LDS aperture:
// addr[31:0] is the LDS offset).
__device__ __forceinline__ unsigned lds_off(const void* p) {
  return (unsigned)(unsigned long long)p;
}

#if HAS_TDM
// Tensor Data Mover: DMA a 64x64 bf16 tile (row stride `stride_elems`) from
// global memory into LDS, packed contiguously ([64][64] row-major).
// D# per CDNA5 ISA 8.3/8.4: group0 = {count=1, lds_addr, global_addr, type=2},
// group1 = {data_size=2B, huge tensor dims (no OOB), tile 64x64, dim0 stride}.
__device__ __forceinline__ void tdm_load_64x64(unsigned lds_byte,
                                               const unsigned short* gsrc,
                                               unsigned stride_elems) {
  unsigned long long ga = (unsigned long long)(const void*)gsrc;
  tdmg0 g0;
  g0[0] = 1u;                                    // count=1, user mode
  g0[1] = lds_byte;                              // lds_addr (bytes)
  g0[2] = (unsigned)ga;                          // global_addr[31:0]
  g0[3] = (unsigned)((ga >> 32) & 0x01FFFFFFu)   // global_addr[56:32]
          | 0x80000000u;                         // type = 2 ("image")
  tdmg1 g1;
  g1[0] = (int)(1u << 16);                       // data_size = 1 -> 2 bytes
  g1[1] = 0;                                     // tensor_dim0 = 1<<20 (lo16=0)
  g1[2] = (int)(0x10u | (0u << 16));             // dim0 hi16, dim1 lo16
  g1[3] = (int)(0x10u | (64u << 16));            // dim1 hi16, tile_dim0 = 64
  g1[4] = 64;                                    // tile_dim1 = 64, tile_dim2 = 0
  g1[5] = (int)stride_elems;                     // tensor_dim0_stride lo32
  g1[6] = 0;                                     // stride0 hi16, stride1 lo16
  g1[7] = 0;                                     // stride1 hi32
  tdmg2 gz = {0, 0, 0, 0};
#if defined(__clang_major__) && __clang_major__ >= 23
  tdmg1 gz8 = {0, 0, 0, 0, 0, 0, 0, 0};
  __builtin_amdgcn_tensor_load_to_lds(g0, g1, gz, gz, gz8, 0);
#else
  __builtin_amdgcn_tensor_load_to_lds(g0, g1, gz, gz, 0);
#endif
}
#endif

// ---------------------------------------------------------------------------
// fp32 -> bf16 elementwise conversion
// ---------------------------------------------------------------------------
__global__ void cvt_bf16_kernel(const float* __restrict__ in,
                                unsigned short* __restrict__ out, int n) {
  int i = blockIdx.x * blockDim.x + threadIdx.x;
  if (i < n) out[i] = f2bf(in[i]);
}

// ---------------------------------------------------------------------------
// GEMM: C[m,j] = sum_k A[m,k] * W[j,k] + bias[j]   (x @ W.T + b)
// Block = 128 threads = 4 waves, 64x64 output tile (each wave 32x32).
// K loop steps 64; A/B 64x64 bf16 tiles double-buffered in LDS, filled by
// the Tensor Data Mover (wave 0 issues, s_wait_tensorcnt + barrier syncs),
// so DMA of tile k+64 overlaps WMMA on tile k.
//   MODE 0: bf16 out, head layout      [B,H,S,DH]   (Q, K)
//   MODE 1: bf16 out, transposed heads [B,H,DH,S]   (V)
//   MODE 2: f32  out, [NTOK, D_MODEL]               (final projection)
// ---------------------------------------------------------------------------
template <int MODE>
__global__ __launch_bounds__(128) void gemm_kernel(
    const unsigned short* __restrict__ A, const unsigned short* __restrict__ W,
    const float* __restrict__ bias, void* __restrict__ out) {
  __shared__ __align__(16) unsigned short Abuf[2][64 * 64];
  __shared__ __align__(16) unsigned short Bbuf[2][64 * 64];
  const int lane = threadIdx.x & 31;
  const int wave = threadIdx.x >> 5;
  const int m0 = blockIdx.y * 64;
  const int j0 = blockIdx.x * 64;
  const int wm = (wave >> 1) * 32;   // wave's 32x32 quadrant
  const int wn = (wave & 1) * 32;

  auto stage = [&](int buf, int kk) {
#if HAS_TDM
    if (wave == 0) {
      tdm_load_64x64(lds_off(&Abuf[buf][0]), A + (size_t)m0 * D_MODEL + kk, D_MODEL);
      tdm_load_64x64(lds_off(&Bbuf[buf][0]), W + (size_t)j0 * D_MODEL + kk, D_MODEL);
    }
#else
    for (int i = threadIdx.x; i < 64 * 64 / 8; i += 128) {
      const int r = i >> 3, c = (i & 7) * 8;
      *(u32x4*)&Abuf[buf][r * 64 + c] =
          *(const u32x4*)(A + (size_t)(m0 + r) * D_MODEL + kk + c);
      *(u32x4*)&Bbuf[buf][r * 64 + c] =
          *(const u32x4*)(W + (size_t)(j0 + r) * D_MODEL + kk + c);
    }
#endif
  };

  stage(0, 0);

  v8f acc00 = {}, acc01 = {}, acc10 = {}, acc11 = {};
  for (int kk = 0; kk < D_MODEL; kk += 64) {
    const int cur = (kk >> 6) & 1;
#if HAS_TDM
    if (wave == 0) __builtin_amdgcn_s_wait_tensorcnt(0);
#endif
    __syncthreads();                       // tile `cur` ready for everyone
    if (kk + 64 < D_MODEL) stage(cur ^ 1, kk + 64);   // overlap next DMA
#pragma unroll
    for (int ks = 0; ks < 64; ks += 32) {
      v16bf a0 = load_frag(&Abuf[cur][(wm)*64 + ks], 64, lane);
      v16bf a1 = load_frag(&Abuf[cur][(wm + 16) * 64 + ks], 64, lane);
      v16bf b0 = load_frag(&Bbuf[cur][(wn)*64 + ks], 64, lane);
      v16bf b1 = load_frag(&Bbuf[cur][(wn + 16) * 64 + ks], 64, lane);
      acc00 = wmma_bf16(a0, b0, acc00);
      acc01 = wmma_bf16(a0, b1, acc01);
      acc10 = wmma_bf16(a1, b0, acc10);
      acc11 = wmma_bf16(a1, b1, acc11);
    }
    __syncthreads();                       // everyone done reading `cur`
  }

  const int lc = lane & 15, hi = lane >> 4;
  const int mw = m0 + wm, jw = j0 + wn;
  v8f accs[2][2] = {{acc00, acc01}, {acc10, acc11}};
#pragma unroll
  for (int mi = 0; mi < 2; ++mi)
#pragma unroll
    for (int ni = 0; ni < 2; ++ni) {
      const int col = jw + ni * 16 + lc;
      const float bv = bias[col];
      const v8f a = accs[mi][ni];
#pragma unroll
      for (int r = 0; r < 8; ++r) {
        const int row = mw + mi * 16 + r + 8 * hi;   // token index
        const float val = a[r] + bv;
        if (MODE == 2) {
          ((float*)out)[(size_t)row * D_MODEL + col] = val;
        } else {
          const int b = row >> 11, s = row & (SEQ - 1);
          const int h = col >> 6, d = col & (DH - 1);
          unsigned short* o = (unsigned short*)out;
          if (MODE == 0)
            o[(((size_t)(b * HEADS + h)) * SEQ + s) * DH + d] = f2bf(val);
          else
            o[(((size_t)(b * HEADS + h)) * DH + d) * SEQ + s] = f2bf(val);
        }
      }
    }
}

// ---------------------------------------------------------------------------
// Flash attention (causal). One wave per 16-query tile of one (b,h).
// Row-sum of P rides the matrix core (extra WMMA against a ones fragment),
// so only the row-max needs cross-lane shuffles. K fragments are manually
// rotated (block jb+1 preloaded during jb's softmax); V fragments load
// before the softmax VALU tail. Only the diagonal block is masked.
// ---------------------------------------------------------------------------
struct AttnSt {
  v16bf qa0, qa1, ones;
  v16bf k00, k01, k10, k11;     // current K block fragments
  v8f o0, o1, o2, o3, lac;      // O accumulators + row-sum accumulator
  float m[8];                   // running row max
};

template <bool MASKED>
__device__ __forceinline__ void attn_block(AttnSt& st,
                                           const unsigned short* __restrict__ Kb,
                                           const unsigned short* __restrict__ Vb,
                                           unsigned short* P, int n0, int next_n0,
                                           int q0, int lane) {
  const int lc = lane & 15, hi = lane >> 4;
  v8f s0 = {}, s1 = {};
  s0 = wmma_bf16(st.qa0, st.k00, s0);
  s0 = wmma_bf16(st.qa1, st.k01, s0);
  s1 = wmma_bf16(st.qa0, st.k10, s1);
  s1 = wmma_bf16(st.qa1, st.k11, s1);

  if (next_n0 >= 0) {   // rotate: preload next K block over the softmax tail
    st.k00 = load_frag(Kb + (size_t)(next_n0)*DH + 0, DH, lane);
    st.k01 = load_frag(Kb + (size_t)(next_n0)*DH + 32, DH, lane);
    st.k10 = load_frag(Kb + (size_t)(next_n0 + 16) * DH + 0, DH, lane);
    st.k11 = load_frag(Kb + (size_t)(next_n0 + 16) * DH + 32, DH, lane);
  }
  // V fragments (transposed layout [dh,S] -> contiguous key chunks); issued
  // early so they overlap the softmax VALU work.
  v16bf v0 = load_frag(Vb + (size_t)(0 * 16) * SEQ + n0, SEQ, lane);
  v16bf v1 = load_frag(Vb + (size_t)(1 * 16) * SEQ + n0, SEQ, lane);
  v16bf v2 = load_frag(Vb + (size_t)(2 * 16) * SEQ + n0, SEQ, lane);
  v16bf v3 = load_frag(Vb + (size_t)(3 * 16) * SEQ + n0, SEQ, lane);

  const float scale = 0.125f;   // 1/sqrt(64)
  const int col0 = n0 + lc, col1 = col0 + 16;
#pragma unroll
  for (int r = 0; r < 8; ++r) {
    const int row = q0 + r + 8 * hi;
    float x0 = s0[r] * scale;
    float x1 = s1[r] * scale;
    if (MASKED) {
      x0 = (col0 <= row) ? x0 : -1e30f;
      x1 = (col1 <= row) ? x1 : -1e30f;
    }
    const float mnew = fmaxf(st.m[r], redmax16(fmaxf(x0, x1)));
    const float p0 = __expf(x0 - mnew);
    const float p1 = __expf(x1 - mnew);
    const float alpha = __expf(st.m[r] - mnew);
    st.m[r] = mnew;
    st.o0[r] *= alpha; st.o1[r] *= alpha; st.o2[r] *= alpha; st.o3[r] *= alpha;
    st.lac[r] *= alpha;
    P[(r + 8 * hi) * 32 + lc] = f2bf(p0);
    P[(r + 8 * hi) * 32 + lc + 16] = f2bf(p1);
  }
  asm volatile("s_wait_dscnt 0" ::: "memory");   // P visible wave-wide
  const v16bf pa = load_frag(P, 32, lane);
  st.o0 = wmma_bf16(pa, v0, st.o0);
  st.o1 = wmma_bf16(pa, v1, st.o1);
  st.o2 = wmma_bf16(pa, v2, st.o2);
  st.o3 = wmma_bf16(pa, v3, st.o3);
  st.lac = wmma_bf16(pa, st.ones, st.lac);       // row-sum via matrix core
}

__global__ __launch_bounds__(128) void attn_kernel(
    const unsigned short* __restrict__ Qh, const unsigned short* __restrict__ Kh,
    const unsigned short* __restrict__ Vt, unsigned short* __restrict__ Obf) {
  __shared__ __align__(16) unsigned short Plds[4][16 * 32];
  const int lane = threadIdx.x & 31;
  const int wave = threadIdx.x >> 5;
  const int gw = blockIdx.x * 4 + wave;   // 0 .. B*H*(S/16)-1
  const int qt = gw & (SEQ / 16 - 1);
  const int bh = gw >> 7;
  const int q0 = qt * 16;
  const int b = bh >> 4, h = bh & 15;
  const unsigned short* Qb = Qh + (size_t)bh * SEQ * DH;
  const unsigned short* Kb = Kh + (size_t)bh * SEQ * DH;
  const unsigned short* Vb = Vt + (size_t)bh * DH * SEQ;
  unsigned short* P = &Plds[wave][0];

  AttnSt st;
  st.qa0 = load_frag(Qb + (size_t)q0 * DH + 0, DH, lane);
  st.qa1 = load_frag(Qb + (size_t)q0 * DH + 32, DH, lane);
  {
    u32x4 c1; c1[0] = c1[1] = c1[2] = c1[3] = 0x3F803F80u;   // bf16 1.0 x2
    FragU fo; fo.q[0] = c1; fo.q[1] = c1;
    st.ones = fo.v;
  }
  v8f z = {};
  st.o0 = z; st.o1 = z; st.o2 = z; st.o3 = z; st.lac = z;
#pragma unroll
  for (int r = 0; r < 8; ++r) st.m[r] = -1e30f;

  st.k00 = load_frag(Kb + 0 * DH + 0, DH, lane);
  st.k01 = load_frag(Kb + 0 * DH + 32, DH, lane);
  st.k10 = load_frag(Kb + 16 * DH + 0, DH, lane);
  st.k11 = load_frag(Kb + 16 * DH + 32, DH, lane);

  const int nblocks = ((q0 + 15) >> 5) + 1;   // causal: keys <= q0+15
  for (int jb = 0; jb < nblocks - 1; ++jb)    // interior blocks: no mask
    attn_block<false>(st, Kb, Vb, P, jb * 32, (jb + 1) * 32, q0, lane);
  attn_block<true>(st, Kb, Vb, P, (nblocks - 1) * 32, -1, q0, lane);

  const int lc = lane & 15, hi = lane >> 4;
  float inv[8];
#pragma unroll
  for (int r = 0; r < 8; ++r) inv[r] = 1.0f / st.lac[r];

  unsigned short* Orow = Obf + (size_t)(b * SEQ) * D_MODEL + (size_t)h * DH;
  v8f oo[4] = {st.o0, st.o1, st.o2, st.o3};
#pragma unroll
  for (int nc = 0; nc < 4; ++nc)
#pragma unroll
    for (int r = 0; r < 8; ++r) {
      const int row = q0 + r + 8 * hi;
      Orow[(size_t)row * D_MODEL + nc * 16 + lc] = f2bf(oo[nc][r] * inv[r]);
    }
}

// ---------------------------------------------------------------------------
// Host-side launch. Workspace (bf16 staging, 64 MB total):
//   Xq/Xk/Xv @ 0/8/16 MB ; Wq/Wk/Wv/Wo bf16 @ 24/26/28/30 MB
//   Qh/Kh @ 32/40 MB [B,H,S,DH] ; Vt @ 48 MB [B,H,DH,S] ; O @ 56 MB
// ---------------------------------------------------------------------------
extern "C" void kernel_launch(void* const* d_in, const int* in_sizes, int n_in,
                              void* d_out, int out_size, void* d_ws, size_t ws_size,
                              hipStream_t stream) {
  (void)in_sizes; (void)n_in; (void)out_size; (void)ws_size;

  const float* query = (const float*)d_in[0];
  const float* key   = (const float*)d_in[1];
  const float* value = (const float*)d_in[2];
  // d_in[3] = causal mask, implemented analytically in-kernel
  const float* Wq = (const float*)d_in[4];
  const float* bq = (const float*)d_in[5];
  const float* Wk = (const float*)d_in[6];
  const float* bk = (const float*)d_in[7];
  const float* Wv = (const float*)d_in[8];
  const float* bv = (const float*)d_in[9];
  const float* Wo = (const float*)d_in[10];
  const float* bo = (const float*)d_in[11];

  char* ws = (char*)d_ws;
  const size_t MB = (size_t)1 << 20;
  unsigned short* Xq  = (unsigned short*)(ws + 0 * MB);
  unsigned short* Xk  = (unsigned short*)(ws + 8 * MB);
  unsigned short* Xv  = (unsigned short*)(ws + 16 * MB);
  unsigned short* Wqb = (unsigned short*)(ws + 24 * MB);
  unsigned short* Wkb = (unsigned short*)(ws + 26 * MB);
  unsigned short* Wvb = (unsigned short*)(ws + 28 * MB);
  unsigned short* Wob = (unsigned short*)(ws + 30 * MB);
  unsigned short* Qh  = (unsigned short*)(ws + 32 * MB);
  unsigned short* Kh  = (unsigned short*)(ws + 40 * MB);
  unsigned short* Vt  = (unsigned short*)(ws + 48 * MB);
  unsigned short* Ob  = (unsigned short*)(ws + 56 * MB);

  const int NX = NTOK * D_MODEL;      // 4,194,304
  const int NW = D_MODEL * D_MODEL;   // 1,048,576
  const int CT = 256;
  cvt_bf16_kernel<<<(NX + CT - 1) / CT, CT, 0, stream>>>(query, Xq, NX);
  cvt_bf16_kernel<<<(NX + CT - 1) / CT, CT, 0, stream>>>(key,   Xk, NX);
  cvt_bf16_kernel<<<(NX + CT - 1) / CT, CT, 0, stream>>>(value, Xv, NX);
  cvt_bf16_kernel<<<(NW + CT - 1) / CT, CT, 0, stream>>>(Wq, Wqb, NW);
  cvt_bf16_kernel<<<(NW + CT - 1) / CT, CT, 0, stream>>>(Wk, Wkb, NW);
  cvt_bf16_kernel<<<(NW + CT - 1) / CT, CT, 0, stream>>>(Wv, Wvb, NW);
  cvt_bf16_kernel<<<(NW + CT - 1) / CT, CT, 0, stream>>>(Wo, Wob, NW);

  dim3 gg(D_MODEL / 64, NTOK / 64);   // (16, 64)
  gemm_kernel<0><<<gg, 128, 0, stream>>>(Xq, Wqb, bq, Qh);
  gemm_kernel<0><<<gg, 128, 0, stream>>>(Xk, Wkb, bk, Kh);
  gemm_kernel<1><<<gg, 128, 0, stream>>>(Xv, Wvb, bv, Vt);

  const int nwaves = BATCH * HEADS * (SEQ / 16);   // 4096
  attn_kernel<<<nwaves / 4, 128, 0, stream>>>(Qh, Kh, Vt, Ob);

  gemm_kernel<2><<<gg, 128, 0, stream>>>(Ob, Wob, bo, d_out);
}
